// PaperGCNModel_20864951124557
// MI455X (gfx1250) — matrix operand order repaired and checked by reading, not compiled, not run on hardware
//
#include <hip/hip_runtime.h>

typedef __attribute__((ext_vector_type(16))) _Float16 v16h;
typedef __attribute__((ext_vector_type(8)))  float    v8f;

#define HB 64  // hidden channels (both GCN layers output 64)

// ---------------------------------------------------------------- utilities
__global__ void fill_k(float* p, float v, int n) {
  int i = blockIdx.x * blockDim.x + threadIdx.x;
  if (i < n) p[i] = v;
}

__global__ void deg_edges_k(float* deg, const int* __restrict__ dst, int E) {
  int e = blockIdx.x * blockDim.x + threadIdx.x;
  if (e < E) atomicAdd(&deg[dst[e]], 1.0f);
}

__global__ void dinv_k(const float* __restrict__ deg, float* dinv, int n) {
  int i = blockIdx.x * blockDim.x + threadIdx.x;
  if (i < n) dinv[i] = rsqrtf(deg[i]);
}

// ------------------------------------------------------- WMMA GEMM: [N,K]@[K,64]
// One wave handles a 16-row tile; 8 waves per block. W is staged into LDS as
// f16 *already packed in WMMA B-fragment order*, so the k-loop loads each
// fragment with one contiguous 32-byte DS read. The per-input-channel affine
// (folded BatchNorm) is a compile-time option so the k-loop stays branch-free.
template <bool AFFINE>
__global__ void gemm_wmma_k(const float* __restrict__ A, const float* __restrict__ W,
                            const float* __restrict__ scale, const float* __restrict__ shift,
                            float* __restrict__ Out, int nrows, int K) {
  __shared__ _Float16 Wh[128 * HB];              // fragment-packed f16, <= 16 KB
  int tid = threadIdx.x;

  // Stage + convert + pack: Wh[((kb*4+ct)*32 + lane)*16 + j] holds the f16
  // element lane/j of B-fragment (kb, ct), per the ISA 32x16 16-bit B layout:
//   lanes 0-15 hold K rows {0..7, 16..23}, lanes 16-31 hold K rows {8..15, 24..31} of the chunk.
  for (int idx = tid; idx < K * HB; idx += blockDim.x) {
    int frag = idx >> 9;                         // kb*4 + ct
    int r    = idx & 511;
    int ln   = r >> 4;                           // lane 0..31
    int j    = r & 15;                           // element within fragment
    int kb   = frag >> 2;
    int ct   = frag & 3;
    int k    = kb * 32 + (ln >> 4) * 8 + (j & 7) + (j >> 3) * 16;
    int col  = ct * 16 + (ln & 15);
    Wh[idx] = (_Float16)W[k * HB + col];
  }
  __syncthreads();

  int wave = tid >> 5;
  int lane = tid & 31;
  int row_base = (blockIdx.x * 8 + wave) * 16;
  if (row_base >= nrows) return;                 // wave-uniform exit, EXEC stays full

  int m  = lane & 15;                            // row within tile (A) / column (B,D)
  int kh = lane >> 4;                            // half-wave selector
  const float* arow = A + (size_t)(row_base + m) * K;

  v8f zero = {0.f, 0.f, 0.f, 0.f, 0.f, 0.f, 0.f, 0.f};
  v8f acc[4];
#pragma unroll
  for (int t = 0; t < 4; ++t) acc[t] = zero;

  int nkb = K >> 5;                              // K in chunks of 32
  for (int kb = 0; kb < nkb; ++kb) {
    // A fragment, ISA 16-bit 16x32 layout:
    //   lanes 0-15 hold K = {0..7, 16..23}; lanes 16-31 hold K = {8..15, 24..31}
    v16h a;
#pragma unroll
    for (int part = 0; part < 2; ++part) {
      int kbase = kb * 32 + kh * 8 + part * 16;
      float4 p0 = *(const float4*)(arow + kbase);
      float4 p1 = *(const float4*)(arow + kbase + 4);
      float v[8] = {p0.x, p0.y, p0.z, p0.w, p1.x, p1.y, p1.z, p1.w};
      if constexpr (AFFINE) {
        float4 s0 = *(const float4*)(scale + kbase);
        float4 s1 = *(const float4*)(scale + kbase + 4);
        float4 h0 = *(const float4*)(shift + kbase);
        float4 h1 = *(const float4*)(shift + kbase + 4);
        float sv[8] = {s0.x, s0.y, s0.z, s0.w, s1.x, s1.y, s1.z, s1.w};
        float hv[8] = {h0.x, h0.y, h0.z, h0.w, h1.x, h1.y, h1.z, h1.w};
#pragma unroll
        for (int jj = 0; jj < 8; ++jj) v[jj] = fmaf(v[jj], sv[jj], hv[jj]);
      }
#pragma unroll
      for (int jj = 0; jj < 8; ++jj) a[part * 8 + jj] = (_Float16)v[jj];
    }
#pragma unroll
    for (int ct = 0; ct < 4; ++ct) {
      v16h b = *(const v16h*)(Wh + ((size_t)(((kb << 2) + ct) * 32 + lane) << 4));
      acc[ct] = __builtin_amdgcn_wmma_f32_16x16x32_f16(
          false, a, false, b, (short)0, acc[ct], false, false);
    }
  }

  // D layout: lanes 0-15 -> M=r, lanes 16-31 -> M=r+8; n = lane&15
#pragma unroll
  for (int ct = 0; ct < 4; ++ct) {
#pragma unroll
    for (int r = 0; r < 8; ++r) {
      int mm = r + kh * 8;
      Out[(size_t)(row_base + mm) * HB + ct * 16 + m] = acc[ct][r];
    }
  }
}

// --------------------------------------------------- GCN normalized aggregation
__global__ void agg_init_k(const float* __restrict__ H, const float* __restrict__ dinv,
                           float* __restrict__ A, long total) {
  long idx = (long)blockIdx.x * blockDim.x + threadIdx.x;
  if (idx < total) {
    int i = (int)(idx >> 6);
    float di = dinv[i];
    A[idx] = H[idx] * di * di;                   // self-loop term
  }
}

__global__ void edge_scatter_k(const float* __restrict__ H, const float* __restrict__ dinv,
                               const int* __restrict__ src, const int* __restrict__ dst,
                               float* A, long totalEF) {
  long idx = (long)blockIdx.x * blockDim.x + threadIdx.x;
  if (idx >= totalEF) return;
  int e = (int)(idx >> 6);
  int f = (int)(idx & 63);
  int s = src[e];
  int d = dst[e];
  float coef = dinv[s] * dinv[d];
  atomicAdd(&A[(size_t)d * HB + f], H[(size_t)s * HB + f] * coef);
}

// ------------------------------------- bias + ReLU + per-channel BN statistics
__global__ void relu_bias_stats_k(float* A, const float* __restrict__ bias,
                                  float* sum, float* sumsq, long total) {
  __shared__ float s1[HB];
  __shared__ float s2[HB];
  int t = threadIdx.x;
  if (t < HB) { s1[t] = 0.f; s2[t] = 0.f; }
  __syncthreads();
  long idx = (long)blockIdx.x * blockDim.x + t;
  if (idx < total) {
    int f = (int)(idx & 63);
    float v = A[idx] + bias[f];
    v = fmaxf(v, 0.f);
    A[idx] = v;
    atomicAdd(&s1[f], v);
    atomicAdd(&s2[f], v * v);
  }
  __syncthreads();
  if (t < HB) {
    atomicAdd(&sum[t], s1[t]);
    atomicAdd(&sumsq[t], s2[t]);
  }
}

__global__ void bn_finalize_k(const float* sum, const float* sq,
                              const float* gamma, const float* beta,
                              float* scale, float* shift, float n) {
  int f = threadIdx.x;
  if (f < HB) {
    float mean = sum[f] / n;
    float var  = sq[f] / n - mean * mean;
    float sc   = gamma[f] * rsqrtf(var + 1e-5f);
    scale[f] = sc;
    shift[f] = beta[f] - mean * sc;
  }
}

// ------------------------------------------- global mean pool (BN2 folded in)
__global__ void pool_k(const float* __restrict__ A, const float* __restrict__ scale,
                       const float* __restrict__ shift, const int* __restrict__ batch,
                       float* gsum, float* gcnt, long total) {
  long idx = (long)blockIdx.x * blockDim.x + threadIdx.x;
  if (idx >= total) return;
  int i = (int)(idx >> 6);
  int f = (int)(idx & 63);
  int g = batch[i];
  float v = A[idx] * scale[f] + shift[f];
  atomicAdd(&gsum[(size_t)g * HB + f], v);
  if (f == 0) atomicAdd(&gcnt[g], 1.0f);
}

// ------------------------------------------------------------- small MLP head
__device__ void dense_layer(const float* in, int din, const float* W, const float* b,
                            float* out, int dout) {
  for (int idx = threadIdx.x; idx < 64 * dout; idx += blockDim.x) {
    int i = idx / dout;
    int o = idx - i * dout;
    float acc = b[o];
    for (int k = 0; k < din; ++k) acc += in[i * din + k] * W[k * dout + o];
    out[idx] = fmaxf(acc, 0.f);
  }
  __syncthreads();
}

__global__ void mlp_k(const float* gsum, const float* gcnt,
                      const float* fW1, const float* fb1,
                      const float* fW2, const float* fb2,
                      const float* fW3, const float* fb3,
                      const float* fW4, const float* fb4,
                      const float* oW, const float* ob, float* out) {
  __shared__ float bA[64 * 128];
  __shared__ float bB[64 * 128];
  int t = threadIdx.x;
  for (int idx = t; idx < 64 * 64; idx += blockDim.x) {
    int i = idx >> 6;
    bA[idx] = gsum[idx] / fmaxf(gcnt[i], 1.0f);
  }
  __syncthreads();
  dense_layer(bA, 64, fW1, fb1, bB, 128);
  dense_layer(bB, 128, fW2, fb2, bA, 64);
  dense_layer(bA, 64, fW3, fb3, bB, 32);
  dense_layer(bB, 32, fW4, fb4, bA, 16);
  if (t < 64) {
    float acc = ob[0];
    for (int k = 0; k < 16; ++k) acc += bA[t * 16 + k] * oW[k];
    out[t] = acc;
  }
}

// ------------------------------------------------------------------- launcher
extern "C" void kernel_launch(void* const* d_in, const int* in_sizes, int n_in,
                              void* d_out, int out_size, void* d_ws, size_t ws_size,
                              hipStream_t stream) {
  const float* x   = (const float*)d_in[0];
  const int*   ei  = (const int*)d_in[1];
  const int*   bat = (const int*)d_in[2];
  const float* W1  = (const float*)d_in[3];
  const float* b1  = (const float*)d_in[4];
  const float* W2  = (const float*)d_in[5];
  const float* b2  = (const float*)d_in[6];
  const float* g1  = (const float*)d_in[7];
  const float* be1 = (const float*)d_in[8];
  const float* g2  = (const float*)d_in[9];
  const float* be2 = (const float*)d_in[10];
  const float* fW1 = (const float*)d_in[11];
  const float* fb1 = (const float*)d_in[12];
  const float* fW2 = (const float*)d_in[13];
  const float* fb2 = (const float*)d_in[14];
  const float* fW3 = (const float*)d_in[15];
  const float* fb3 = (const float*)d_in[16];
  const float* fW4 = (const float*)d_in[17];
  const float* fb4 = (const float*)d_in[18];
  const float* oW  = (const float*)d_in[19];
  const float* ob  = (const float*)d_in[20];

  const int N = in_sizes[2];
  const int E = in_sizes[1] / 2;
  const int* src = ei;
  const int* dst = ei + E;

  // ---- workspace layout (floats) ----
  float* ws     = (float*)d_ws;
  float* deg    = ws;                         // N
  float* dinv   = ws + (size_t)N;             // N
  float* stats  = ws + 2 * (size_t)N;         // 512
  float* sum1   = stats,        *sq1    = stats + 64;
  float* sum2   = stats + 128,  *sq2    = stats + 192;
  float* scale1 = stats + 256,  *shift1 = stats + 320;
  float* scale2 = stats + 384,  *shift2 = stats + 448;
  float* gsum   = stats + 512;                // 64*64
  float* gcnt   = gsum + 64 * 64;             // 64
  float* bufH   = gcnt + 64;                  // N*64
  float* bufA   = bufH + (size_t)N * HB;      // N*64

  const int B = 256;
  const long NF = (long)N * HB;
  const long EF = (long)E * HB;
  const int nBlk  = (N + B - 1) / B;
  const int eBlk  = (E + B - 1) / B;
  const int nfBlk = (int)((NF + B - 1) / B);
  const int efBlk = (int)((EF + B - 1) / B);
  const int gemmBlk = (((N + 15) / 16) + 7) / 8;

  // degree (with self-loop) and D^{-1/2}; zero accumulators
  fill_k<<<nBlk, B, 0, stream>>>(deg, 1.0f, N);
  fill_k<<<(4672 + B - 1) / B, B, 0, stream>>>(stats, 0.0f, 512 + 64 * 64 + 64);
  deg_edges_k<<<eBlk, B, 0, stream>>>(deg, dst, E);
  dinv_k<<<nBlk, B, 0, stream>>>(deg, dinv, N);

  // ---- layer 1: H1 = X @ W1 (WMMA), normalized aggregation, bias+ReLU+BN stats
  gemm_wmma_k<false><<<gemmBlk, 256, 0, stream>>>(x, W1, nullptr, nullptr, bufH, N, 128);
  agg_init_k<<<nfBlk, B, 0, stream>>>(bufH, dinv, bufA, NF);
  edge_scatter_k<<<efBlk, B, 0, stream>>>(bufH, dinv, src, dst, bufA, EF);
  relu_bias_stats_k<<<nfBlk, B, 0, stream>>>(bufA, b1, sum1, sq1, NF);
  bn_finalize_k<<<1, 64, 0, stream>>>(sum1, sq1, g1, be1, scale1, shift1, (float)N);

  // ---- layer 2: BN1 folded into A-load of GEMM2
  gemm_wmma_k<true><<<gemmBlk, 256, 0, stream>>>(bufA, W2, scale1, shift1, bufH, N, HB);
  agg_init_k<<<nfBlk, B, 0, stream>>>(bufH, dinv, bufA, NF);
  edge_scatter_k<<<efBlk, B, 0, stream>>>(bufH, dinv, src, dst, bufA, EF);
  relu_bias_stats_k<<<nfBlk, B, 0, stream>>>(bufA, b2, sum2, sq2, NF);
  bn_finalize_k<<<1, 64, 0, stream>>>(sum2, sq2, g2, be2, scale2, shift2, (float)N);

  // ---- pool (BN2 folded in) + MLP head
  pool_k<<<nfBlk, B, 0, stream>>>(bufA, scale2, shift2, bat, gsum, gcnt, NF);
  mlp_k<<<1, 256, 0, stream>>>(gsum, gcnt, fW1, fb1, fW2, fb2, fW3, fb3, fW4, fb4,
                               oW, ob, (float*)d_out);
}
